// Scaled_DotProduct_Attention_22033182228928
// MI455X (gfx1250) — matrix operand order repaired and checked by reading, not compiled, not run on hardware
//
#include <hip/hip_runtime.h>
#include <hip/hip_bf16.h>

// Problem constants (reference: B,H,S,D = 4,16,2048,64)
#define BATCH 4
#define HEADS 16
#define SEQ   2048
#define HDIM  64
#define KT    32     // keys per tile (WMMA K depth for P@V)
#define QT    16     // query rows per block (WMMA M)

typedef __attribute__((ext_vector_type(16))) __bf16 v16bf;
typedef __attribute__((ext_vector_type(8)))  float  v8f;

union BF16x16 { unsigned int u[8]; v16bf v; };

// pack two f32 -> two bf16 (round-to-nearest-even) in one dword
__device__ __forceinline__ unsigned int pack_bf16(float a, float b) {
  unsigned int ua = __float_as_uint(a);
  unsigned int ub = __float_as_uint(b);
  ua += 0x7FFFu + ((ua >> 16) & 1u);
  ub += 0x7FFFu + ((ub >> 16) & 1u);
  return (ua >> 16) | (ub & 0xFFFF0000u);
}
__device__ __forceinline__ unsigned short bf16_1(float a) {
  unsigned int ua = __float_as_uint(a);
  ua += 0x7FFFu + ((ua >> 16) & 1u);
  return (unsigned short)(ua >> 16);
}

// ISA 16-bit A-matrix 16x32 layout: lanes0-15: V0-3 => K=0..7, V4-7 => K=16..23
//                                   lanes16-31: V0-3 => K=8..15, V4-7 => K=24..31
__device__ __forceinline__ int a_kidx(int r, int hi) {
  return (r < 4) ? (8 * hi + 2 * r) : (16 + 8 * hi + 2 * (r - 4));
}
// B-matrix 32x16 layout matching A: lanes0-15: V0-3 => K=0..7, V4-7 => K=16..23
//                                  lanes16-31: V0-3 => K=8..15, V4-7 => K=24..31
__device__ __forceinline__ int b_kidx(int r, int hi) {
  return (r < 4) ? (8 * hi + 2 * r) : (16 + 8 * hi + 2 * (r - 4));
}

// LDS layout offsets (bytes)
#define SK_OFF   0                      // [H][32 keys][64 d] bf16   = 64 KB
#define SVT_OFF  65536                  // [H][64 d][32 keys] bf16   = 64 KB
#define SS_OFF   131072                 // [H][16 q][32 k] f32       = 32 KB
#define SP_OFF   (131072 + 32768)       // [H][16 q][32 k] bf16      = 16 KB
#define SM_OFF   (131072 + 32768 + 16384) // [H][16 q * 32 k] u8     =  8 KB
#define LDS_TOTAL (SM_OFF + 8192)       // 184 KB

__global__ void __launch_bounds__(512, 1)
sdpa_headsoftmax_kernel(const float* __restrict__ Q,
                        const float* __restrict__ K,
                        const float* __restrict__ V,
                        const unsigned char* __restrict__ mask,
                        float* __restrict__ out) {
  extern __shared__ char smem_raw[];
  unsigned int*   sK  = (unsigned int*)(smem_raw + SK_OFF);   // dword view
  unsigned short* sVT = (unsigned short*)(smem_raw + SVT_OFF);
  float*          sS  = (float*)(smem_raw + SS_OFF);
  unsigned short* sP  = (unsigned short*)(smem_raw + SP_OFF);
  unsigned char*  sM  = (unsigned char*)(smem_raw + SM_OFF);

  const int tid  = threadIdx.x;
  const int lane = tid & 31;
  const int h    = tid >> 5;       // one wave per head
  const int m16  = lane & 15;
  const int hi   = lane >> 4;

  const int blk = blockIdx.x;
  const int b   = blk >> 7;            // SEQ/QT = 128 q-tiles per batch
  const int q0  = (blk & 127) << 4;

  const size_t headBase = ((size_t)(b * HEADS + h)) * SEQ * HDIM;
  const float* Qh = Q + headBase + (size_t)q0 * HDIM;
  const float* Kh = K + headBase;
  const float* Vh = V + headBase;
  const unsigned char* Mh = mask + ((size_t)(b * HEADS + h) * SEQ + q0) * SEQ;

  // ---- Preload Q tile as two 16x32 bf16 WMMA A operands (d 0..31, 32..63)
  BF16x16 aQ[2];
#pragma unroll
  for (int c = 0; c < 2; ++c)
#pragma unroll
    for (int r = 0; r < 8; ++r) {
      int d = c * 32 + a_kidx(r, hi);
      float f0 = Qh[m16 * HDIM + d];
      float f1 = Qh[m16 * HDIM + d + 1];
      aQ[c].u[r] = pack_bf16(f0, f1);
    }

  const float scale = 0.125f;  // 1/sqrt(64)
  v8f zero = {0, 0, 0, 0, 0, 0, 0, 0};
  v8f ctx[4];
#pragma unroll
  for (int dt = 0; dt < 4; ++dt) ctx[dt] = zero;

  unsigned int*   sKh  = sK  + h * (KT * HDIM / 2);
  unsigned short* sVTh = sVT + h * (HDIM * KT);
  float*          sSh  = sS  + h * (QT * KT);
  unsigned short* sPh  = sP  + h * (QT * KT);
  unsigned char*  sMh  = sM  + h * (QT * KT);

  // Per-lane mask staging coordinates: one 16B chunk per lane covers the
  // head's whole 16x32 byte tile (16 rows x 2 halves).
  const int mrow  = lane >> 1;
  const int mhalf = (lane & 1) << 4;   // 0 or 16
  const unsigned int mLdsOff =
      (unsigned int)(SM_OFF + h * (QT * KT) + mrow * KT + mhalf);

  for (int k0 = 0; k0 < SEQ; k0 += KT) {
    // ---- Async DMA this head's mask tile (16x32 bytes) into LDS.
    //      Raw byte copy -> perfect fit for global_load_async_to_lds_b128;
    //      lands while we do the f32->bf16 K/V staging below.
    {
      unsigned long long gaddr =
          (unsigned long long)(Mh + (size_t)mrow * SEQ + k0 + mhalf);
      asm volatile("global_load_async_to_lds_b128 %0, %1, off"
                   :: "v"(mLdsOff), "v"(gaddr)
                   : "memory");
    }

    // prefetch next tile's K/V into near caches (global_prefetch_b8)
    if (k0 + KT < SEQ) {
      __builtin_prefetch(Kh + (size_t)(k0 + KT) * HDIM + lane * 64, 0, 3);
      __builtin_prefetch(Vh + (size_t)(k0 + KT) * HDIM + lane * 64, 0, 3);
    }

    // ---- Stage this head's K tile (bf16 row-major) and V^T tile into LDS.
    //      Wave-private data: DS ops in-order within a wave, no barrier needed.
#pragma unroll
    for (int i = 0; i < 16; ++i) {
      int idx = i * 32 + lane;          // 512 float4s cover 32x64 tile
      int key = idx >> 4;
      int dg  = (idx & 15) << 2;        // d in groups of 4
      const float4 kv = *(const float4*)(Kh + (size_t)(k0 + key) * HDIM + dg);
      sKh[(key * HDIM + dg) >> 1]       = pack_bf16(kv.x, kv.y);
      sKh[((key * HDIM + dg) >> 1) + 1] = pack_bf16(kv.z, kv.w);
      const float4 vv = *(const float4*)(Vh + (size_t)(k0 + key) * HDIM + dg);
      sVTh[(dg + 0) * KT + key] = bf16_1(vv.x);
      sVTh[(dg + 1) * KT + key] = bf16_1(vv.y);
      sVTh[(dg + 2) * KT + key] = bf16_1(vv.z);
      sVTh[(dg + 3) * KT + key] = bf16_1(vv.w);
    }

    // mask DMA must have landed before we read it below
    asm volatile("s_wait_asynccnt 0" ::: "memory");

    // ---- Scores: S[16q x 32k] = (Q K^T) * scale, masked -> LDS (f32)
#pragma unroll
    for (int nt = 0; nt < 2; ++nt) {
      v8f acc = zero;
#pragma unroll
      for (int c = 0; c < 2; ++c) {
        BF16x16 bK;
#pragma unroll
        for (int r = 0; r < 8; ++r) {
          int d = c * 32 + b_kidx(r, hi);            // reduction dim = d
          bK.u[r] = sKh[((nt * 16 + m16) * HDIM + d) >> 1];
        }
        acc = __builtin_amdgcn_wmma_f32_16x16x32_bf16(
            false, aQ[c].v, false, bK.v, (short)0, acc, false, false);
      }
#pragma unroll
      for (int v = 0; v < 8; ++v) {
        int qrow = v + 8 * hi;                       // C/D layout: VGPRv, half-wave
        int kcol = nt * 16 + m16;
        unsigned char mb = sMh[qrow * KT + kcol];    // ds_load_u8 from staged mask
        sSh[qrow * KT + kcol] = mb ? -__builtin_inff() : acc[v] * scale;
      }
    }
    __syncthreads();

    // ---- Cross-head softmax: one thread per (q,k) pair (512 pairs = blockDim)
    {
      int qrow = tid >> 5;
      int kcol = tid & 31;
      float x[HEADS];
      float mmax = -__builtin_inff();
#pragma unroll
      for (int hh = 0; hh < HEADS; ++hh) {
        x[hh] = sS[hh * (QT * KT) + qrow * KT + kcol];
        mmax = fmaxf(mmax, x[hh]);
      }
      if (mmax == -__builtin_inff()) {
#pragma unroll
        for (int hh = 0; hh < HEADS; ++hh)
          sP[hh * (QT * KT) + qrow * KT + kcol] = 0;
      } else {
        float ssum = 0.f;
#pragma unroll
        for (int hh = 0; hh < HEADS; ++hh) {
          x[hh] = __expf(x[hh] - mmax);
          ssum += x[hh];
        }
        float rinv = __builtin_amdgcn_rcpf(ssum);
#pragma unroll
        for (int hh = 0; hh < HEADS; ++hh)
          sP[hh * (QT * KT) + qrow * KT + kcol] = bf16_1(x[hh] * rinv);
      }
    }
    __syncthreads();

    // ---- ctx += P @ V : A = P (16x32 bf16), B = V^T 32x16 chunks, 4 d-tiles
    BF16x16 aP;
#pragma unroll
    for (int r = 0; r < 8; ++r) {
      int kk = a_kidx(r, hi);
      aP.u[r] = ((const unsigned int*)sPh)[(m16 * KT + kk) >> 1];
    }
#pragma unroll
    for (int dt = 0; dt < 4; ++dt) {
      BF16x16 bV;
#pragma unroll
      for (int r = 0; r < 8; ++r) {
        int kk = b_kidx(r, hi);                      // reduction dim = key
        int d  = dt * 16 + m16;
        bV.u[r] = ((const unsigned int*)sVTh)[(d * KT + kk) >> 1];
      }
      ctx[dt] = __builtin_amdgcn_wmma_f32_16x16x32_bf16(
          false, aP.v, false, bV.v, (short)0, ctx[dt], false, false);
    }
    // No trailing barrier needed: next iteration's sS/sP writers are all
    // behind the next __syncthreads() pair relative to this wave's reads,
    // and sM is wave-private (guarded by s_wait_asynccnt within the wave).
  }

  // ---- Epilogue: ctx (f32) -> out [B, S, H, D]
#pragma unroll
  for (int dt = 0; dt < 4; ++dt)
#pragma unroll
    for (int v = 0; v < 8; ++v) {
      int qrow = v + 8 * hi;
      int d    = dt * 16 + m16;
      out[(((size_t)(b * SEQ + q0 + qrow)) * HEADS + h) * HDIM + d] = ctx[dt][v];
    }
}

extern "C" void kernel_launch(void* const* d_in, const int* in_sizes, int n_in,
                              void* d_out, int out_size, void* d_ws, size_t ws_size,
                              hipStream_t stream) {
  // setup_inputs order: Q, K, V, head_dim(scalar), mask_bool
  const float* Q = (const float*)d_in[0];
  const float* K = (const float*)d_in[1];
  const float* V = (const float*)d_in[2];
  const unsigned char* mask = (const unsigned char*)d_in[4];  // jax bool = 1 byte
  float* out = (float*)d_out;

  dim3 grid(BATCH * (SEQ / QT));   // 512 blocks: one (batch, 16-query tile) each
  dim3 block(512);                 // 16 waves = 16 heads
  size_t lds = LDS_TOTAL;          // 184 KB dynamic LDS
  hipLaunchKernelGGL(sdpa_headsoftmax_kernel, grid, block, lds, stream,
                     Q, K, V, mask, out);
}
